// GMAEModel_17489106829550
// MI455X (gfx1250) — matrix operand order repaired and checked by reading, not compiled, not run on hardware
//
#include <hip/hip_runtime.h>
#include <hip/hip_bf16.h>
#include <math.h>

// ---------------------------------------------------------------------------
// GraphMAE GAT autoencoder for MI455X (gfx1250, wave32, WMMA).
// Dense GEMMs: v_wmma_f32_16x16x32_bf16, one wave per SUBx16 output panel
// (SUB=4 main grid, SUB=1..3 tail instantiations; compile-time SUB keeps the
// k-loop branch-free so accumulators stay pinned -> no spills).
// Edge softmax + aggregation via L2-resident float atomics (node tables are
// 25.6MB << 192MB L2).
// ---------------------------------------------------------------------------

#define NN      50000
#define NE      800000
#define NDIM    64
#define EDIM    32
#define HID     128
#define HEADS   4
#define DHH     32
#define NMASK_  25000
#define NSAMP_  10000
#define NEG_SLOPE 0.2f

typedef __attribute__((ext_vector_type(16))) __bf16 v16bf;
typedef __attribute__((ext_vector_type(8)))  float  v8f;

// ---------------------------------------------------------------------------
// device helpers
// ---------------------------------------------------------------------------
__device__ __forceinline__ float warp_sum(float v) {
  #pragma unroll
  for (int o = 16; o > 0; o >>= 1) v += __shfl_xor(v, o, 32);
  return v;
}

__device__ __forceinline__ void atomicMaxFloat(float* addr, float val) {
  // bit-trick float max; valid with -inf initialization
  if (val >= 0.0f) atomicMax((int*)addr, __float_as_int(val));
  else             atomicMin((unsigned int*)addr, __float_as_uint(val));
}

// ---------------------------------------------------------------------------
// WMMA GEMM: D[M,N] = act( A[M,K] @ W[K,N] + Cinit + bias )
// One wave computes a (SUB*16)x16 output panel; SUB m-subtiles share one B
// fragment per 32-deep k-step. SUB is a template parameter so the k-loop is
// straight-line (EXEC all-1s, accumulators pinned in contiguous VGPR tuples).
// Fragment layouts (wave32):
//  A 16x32 bf16: lane L (half=L>>4, m=L&15): elem j -> K=half*16+j
//  B 32x16 bf16: lane L: col n=L&15, elem j -> K=half*16+j
//  C/D 16x16 f32: lane L, vgpr v -> D[v + 8*half][L&15]
// ---------------------------------------------------------------------------
template <int SUB>
__global__ void wmma_gemm_kernel(const float* __restrict__ A, int lda,
                                 const float* __restrict__ W, int ldw,
                                 float* D, int ldd,
                                 const float* Cinit,             // may be null (ld = ldd)
                                 const float* __restrict__ bias, // may be null
                                 int m_base, int K, int act) {
  const int lane = threadIdx.x & 31;
  const int half = lane >> 4;
  const int l16  = lane & 15;
  const int m0 = m_base + blockIdx.x * (SUB * 16);
  const int n0 = blockIdx.y * 16;
  const int bn = n0 + l16;

  v8f acc[SUB];
  #pragma unroll
  for (int s = 0; s < SUB; ++s)
    #pragma unroll
    for (int i = 0; i < 8; ++i) acc[s][i] = 0.0f;

  for (int k = 0; k < K; k += 32) {
    // B fragment: shared across the SUB m-subtiles
    v16bf b;
    const float* wcol = W + (size_t)(k + half * 16) * ldw + bn;
    #pragma unroll
    for (int t = 0; t < 16; ++t) b[t] = (__bf16)wcol[(size_t)t * ldw];

    #pragma unroll
    for (int s = 0; s < SUB; ++s) {
      const float* arow = A + (size_t)(m0 + s * 16 + l16) * lda + k + half * 16;
      const float4 p0 = *(const float4*)(arow);
      const float4 p1 = *(const float4*)(arow + 4);
      const float4 p2 = *(const float4*)(arow + 8);
      const float4 p3 = *(const float4*)(arow + 12);
      v16bf a;
      a[0]  = (__bf16)p0.x; a[1]  = (__bf16)p0.y; a[2]  = (__bf16)p0.z; a[3]  = (__bf16)p0.w;
      a[4]  = (__bf16)p1.x; a[5]  = (__bf16)p1.y; a[6]  = (__bf16)p1.z; a[7]  = (__bf16)p1.w;
      a[8]  = (__bf16)p2.x; a[9]  = (__bf16)p2.y; a[10] = (__bf16)p2.z; a[11] = (__bf16)p2.w;
      a[12] = (__bf16)p3.x; a[13] = (__bf16)p3.y; a[14] = (__bf16)p3.z; a[15] = (__bf16)p3.w;
      acc[s] = __builtin_amdgcn_wmma_f32_16x16x32_bf16(
          /*neg_a=*/false, a, /*neg_b=*/false, b,
          /*c_mod=*/(short)0, acc[s], /*reuse_a=*/false, /*reuse_b=*/false);
    }
  }

  // epilogue: wave-uniform feature selection
  const float badd = bias ? bias[bn] : 0.0f;
  if (Cinit != nullptr) {
    #pragma unroll
    for (int s = 0; s < SUB; ++s) {
      const int mb = m0 + s * 16 + 8 * half;
      #pragma unroll
      for (int v = 0; v < 8; ++v) {
        float val = acc[s][v] + badd + Cinit[(size_t)(mb + v) * ldd + bn];
        if (act == 1) val = val > 0.0f ? val : NEG_SLOPE * val;
        D[(size_t)(mb + v) * ldd + bn] = val;
      }
    }
  } else {
    #pragma unroll
    for (int s = 0; s < SUB; ++s) {
      const int mb = m0 + s * 16 + 8 * half;
      #pragma unroll
      for (int v = 0; v < 8; ++v) {
        float val = acc[s][v] + badd;
        if (act == 1) val = val > 0.0f ? val : NEG_SLOPE * val;
        D[(size_t)(mb + v) * ldd + bn] = val;
      }
    }
  }
}

// host-side dispatcher: full 64-row panels with SUB=4, tail with SUB=1..3
static void launch_gemm(const float* A, int lda, const float* W, int ldw,
                        float* D, int ldd, const float* Cinit, const float* bias,
                        int M, int K, int Nout, int act, hipStream_t stream) {
  const int full = M / 64;
  const int rem  = (M % 64) / 16;   // M is a multiple of 16
  const dim3 blk(32);
  if (full > 0)
    wmma_gemm_kernel<4><<<dim3(full, Nout / 16), blk, 0, stream>>>(
        A, lda, W, ldw, D, ldd, Cinit, bias, 0, K, act);
  const int mb = full * 64;
  if (rem == 1)
    wmma_gemm_kernel<1><<<dim3(1, Nout / 16), blk, 0, stream>>>(
        A, lda, W, ldw, D, ldd, Cinit, bias, mb, K, act);
  else if (rem == 2)
    wmma_gemm_kernel<2><<<dim3(1, Nout / 16), blk, 0, stream>>>(
        A, lda, W, ldw, D, ldd, Cinit, bias, mb, K, act);
  else if (rem == 3)
    wmma_gemm_kernel<3><<<dim3(1, Nout / 16), blk, 0, stream>>>(
        A, lda, W, ldw, D, ldd, Cinit, bias, mb, K, act);
}

// ---------------------------------------------------------------------------
// small kernels
// ---------------------------------------------------------------------------
__global__ void copy_f32(const float* __restrict__ in, float* __restrict__ out, long n) {
  long i = (long)blockIdx.x * blockDim.x + threadIdx.x;
  if (i < n) out[i] = in[i];
}

__global__ void apply_mask_token(const int* __restrict__ mask_nodes,
                                 const float* __restrict__ tok, float* xm) {
  long i = (long)blockIdx.x * blockDim.x + threadIdx.x;
  if (i >= (long)NMASK_ * NDIM) return;
  int node = mask_nodes[i / NDIM];
  int d = (int)(i % NDIM);
  xm[(size_t)node * NDIM + d] = tok[d];
}

__global__ void zero_acc(float* acc) {
  if (threadIdx.x < 4) acc[threadIdx.x] = 0.0f;
}

// Wcomb[k,h] = sum_d We[k, h*dh+d] * ae[h,d]   (collapses the fe tensor)
__global__ void make_wcomb(const float* __restrict__ We, int ldwe,
                           const float* __restrict__ ae, int heads, int dh,
                           float* __restrict__ wcomb) {
  int idx = blockIdx.x * blockDim.x + threadIdx.x;
  if (idx >= EDIM * heads) return;
  int k = idx / heads, h = idx % heads;
  float s = 0.0f;
  for (int d = 0; d < dh; ++d) s += We[(size_t)k * ldwe + h * dh + d] * ae[h * dh + d];
  wcomb[idx] = s;
}

// el/er per node-head; also init segment max/sum buffers
__global__ void node_attn_init(const float* __restrict__ feat, int ldf,
                               const float* __restrict__ al, const float* __restrict__ ar,
                               int heads, int dh,
                               float* el, float* er, float* m, float* s, int n) {
  int idx = blockIdx.x * blockDim.x + threadIdx.x;
  if (idx >= n * heads) return;
  int node = idx / heads, h = idx % heads;
  const float* f = feat + (size_t)node * ldf + h * dh;
  float a = 0.0f, b = 0.0f;
  for (int d = 0; d < dh; ++d) { a += f[d] * al[h * dh + d]; b += f[d] * ar[h * dh + d]; }
  el[idx] = a; er[idx] = b;
  m[idx] = -__builtin_inff();
  s[idx] = 0.0f;
}

// logits = leakyrelu(el[src] + er[dst] + edge_attr@wcomb); segment-max into m
__global__ void edge_logits_max(const float* __restrict__ ea,
                                const float* __restrict__ wcomb,
                                const float* __restrict__ el, const float* __restrict__ er,
                                const int* __restrict__ src, const int* __restrict__ dst,
                                int heads, float* logits, float* m) {
  int idx = blockIdx.x * blockDim.x + threadIdx.x;
  if (idx >= NE * heads) return;
  int e = idx / heads, h = idx % heads;
  const float* row = ea + (size_t)e * EDIM;
  float ee = 0.0f;
  for (int k = 0; k < EDIM; ++k) ee += row[k] * wcomb[k * heads + h];
  float lg = el[src[e] * heads + h] + er[dst[e] * heads + h] + ee;
  lg = lg > 0.0f ? lg : NEG_SLOPE * lg;
  logits[idx] = lg;
  atomicMaxFloat(&m[dst[e] * heads + h], lg);
}

// ex = exp(logit - m[dst]) stored in-place; segment-sum into s
__global__ void edge_exp_sum(float* logits, const float* __restrict__ m,
                             float* s, const int* __restrict__ dst, int heads) {
  int idx = blockIdx.x * blockDim.x + threadIdx.x;
  if (idx >= NE * heads) return;
  int e = idx / heads, h = idx % heads;
  float ex = __expf(logits[idx] - m[dst[e] * heads + h]);
  logits[idx] = ex;
  atomicAdd(&s[dst[e] * heads + h], ex);
}

// out[dst, d] += feat[src, d] * ex/s[dst]  (out pre-initialized with residual)
// 4 contiguous dims per thread: one float4 gather + 1 alpha per 4 atomics.
__global__ void scatter_agg(const float* __restrict__ feat, const float* __restrict__ ex,
                            const float* __restrict__ s,
                            const int* __restrict__ src, const int* __restrict__ dst,
                            int heads, int dh, float* out) {
  const int HD = heads * dh;
  const int Q = HD >> 2;
  long idx = (long)blockIdx.x * blockDim.x + threadIdx.x;
  if (idx >= (long)NE * Q) return;
  int e  = (int)(idx / Q);
  int d4 = (int)(idx % Q) << 2;
  int h  = d4 / dh;                 // dh is a multiple of 4 -> same head for all 4
  int dn = dst[e], sn = src[e];
  float a = ex[(size_t)e * heads + h] / s[(size_t)dn * heads + h];
  const float4 f = *(const float4*)(feat + (size_t)sn * HD + d4);
  float* o = out + (size_t)dn * HD + d4;
  atomicAdd(o + 0, f.x * a);
  atomicAdd(o + 1, f.y * a);
  atomicAdd(o + 2, f.z * a);
  atomicAdd(o + 3, f.w * a);
}

// per-node LayerNorm over 128 dims + ReLU, one wave32 per node
__global__ void ln_relu(float* out, const float* __restrict__ gamma,
                        const float* __restrict__ beta, int n) {
  int wave = (int)(((long)blockIdx.x * blockDim.x + threadIdx.x) >> 5);
  int lane = threadIdx.x & 31;
  if (wave >= n) return;
  float* row = out + (size_t)wave * HID;
  float v[4], sum = 0.0f, sq = 0.0f;
  #pragma unroll
  for (int j = 0; j < 4; ++j) { v[j] = row[lane + 32 * j]; sum += v[j]; sq += v[j] * v[j]; }
  sum = warp_sum(sum); sq = warp_sum(sq);
  float mu = sum * (1.0f / HID);
  float var = sq * (1.0f / HID) - mu * mu;
  float rs = rsqrtf(var + 1e-5f);
  #pragma unroll
  for (int j = 0; j < 4; ++j) {
    int c = lane + 32 * j;
    float o = (v[j] - mu) * rs * gamma[c] + beta[c];
    row[c] = o > 0.0f ? o : 0.0f;
  }
}

// SCE: mean over masked nodes of (1 - cos(recon, x))^2 ; one wave per node
__global__ void sce_loss(const float* __restrict__ recon, const float* __restrict__ x,
                         const int* __restrict__ mask, float* acc) {
  int wave = (int)(((long)blockIdx.x * blockDim.x + threadIdx.x) >> 5);
  int lane = threadIdx.x & 31;
  if (wave >= NMASK_) return;
  int node = mask[wave];
  float dot = 0.0f, nr = 0.0f, ni = 0.0f;
  #pragma unroll
  for (int j = 0; j < 2; ++j) {
    int d = lane + 32 * j;
    float r = recon[(size_t)node * NDIM + d];
    float xi = x[(size_t)node * NDIM + d];
    dot += r * xi; nr += r * r; ni += xi * xi;
  }
  dot = warp_sum(dot); nr = warp_sum(nr); ni = warp_sum(ni);
  if (lane == 0) {
    float c = 1.0f - dot / ((sqrtf(nr) + 1e-8f) * (sqrtf(ni) + 1e-8f));
    atomicAdd(acc, c * c);
  }
}

// build [2*NSAMP, 6*HID] = concat(enc_rep[s_idx], enc_rep[d_idx]); enc_rep = [h1|h2|h3]
__global__ void gather_featcat(const float* __restrict__ h1, const float* __restrict__ h2,
                               const float* __restrict__ h3,
                               const int* __restrict__ src, const int* __restrict__ dst,
                               const int* __restrict__ pos, const int* __restrict__ nsrc,
                               const int* __restrict__ ndst, float* __restrict__ featcat) {
  long idx = (long)blockIdx.x * blockDim.x + threadIdx.x;
  const long total = (long)2 * NSAMP_ * (6 * HID);
  if (idx >= total) return;
  int r = (int)(idx / (6 * HID));
  int c = (int)(idx % (6 * HID));
  int node;
  if (c < 3 * HID) {
    node = (r < NSAMP_) ? src[pos[r]] : nsrc[r - NSAMP_];
  } else {
    node = (r < NSAMP_) ? dst[pos[r]] : ndst[r - NSAMP_];
    c -= 3 * HID;
  }
  const float* h = (c < HID) ? h1 : (c < 2 * HID ? h2 : h3);
  featcat[idx] = h[(size_t)node * HID + (c % HID)];
}

// BCE head: pred = sigmoid(hfc @ fc2_W + b); one wave per sample
__global__ void bce_loss(const float* __restrict__ hfc, const float* __restrict__ fc2W,
                         const float* __restrict__ fc2b, float* acc) {
  int wave = (int)(((long)blockIdx.x * blockDim.x + threadIdx.x) >> 5);
  int lane = threadIdx.x & 31;
  if (wave >= 2 * NSAMP_) return;
  float dot = 0.0f;
  #pragma unroll
  for (int j = 0; j < 4; ++j) {
    int c = lane + 32 * j;
    dot += hfc[(size_t)wave * HID + c] * fc2W[c];
  }
  dot = warp_sum(dot);
  if (lane == 0) {
    float z = dot + fc2b[0];
    float p = 1.0f / (1.0f + __expf(-z));
    p = fminf(fmaxf(p, 1e-7f), 1.0f - 1e-7f);
    float loss = (wave < NSAMP_) ? -__logf(p) : -__logf(1.0f - p);
    atomicAdd(acc + 1, loss);
  }
}

__global__ void finalize(const float* __restrict__ acc, float* out) {
  if (threadIdx.x == 0)
    out[0] = acc[0] * (1.0f / NMASK_) + acc[1] * (1.0f / (2.0f * NSAMP_));
}

// ---------------------------------------------------------------------------
// host launcher
// ---------------------------------------------------------------------------
static inline dim3 blks(long n, int t) { return dim3((unsigned)((n + t - 1) / t)); }

extern "C" void kernel_launch(void* const* d_in, const int* in_sizes, int n_in,
                              void* d_out, int out_size, void* d_ws, size_t ws_size,
                              hipStream_t stream) {
  (void)in_sizes; (void)n_in; (void)out_size; (void)ws_size;
  // Input order: jax pytree flatten of setup_inputs():
  //   params leaves first (sorted dict keys recursively):
  //   0 W_e2d | 1..8 dec{W,We,Wres,ae,al,ar,beta,gamma} |
  //   9..32 enc[0..2]{W,We,Wres,ae,al,ar,beta,gamma} |
  //   33 fc1_W 34 fc1_b 35 fc2_W 36 fc2_b 37 mask_token |
  //   then 38 x 39 edge_attr 40 src 41 dst 42 mask_nodes 43 pos_idx 44 neg_src 45 neg_dst
  const float* W_e2d = (const float*)d_in[0];
  const float* decW    = (const float*)d_in[1];
  const float* decWe   = (const float*)d_in[2];
  const float* decWres = (const float*)d_in[3];
  const float* decAe   = (const float*)d_in[4];
  const float* decAl   = (const float*)d_in[5];
  const float* decAr   = (const float*)d_in[6];
  const float* encW[3]    = {(const float*)d_in[9],  (const float*)d_in[17], (const float*)d_in[25]};
  const float* encWe[3]   = {(const float*)d_in[10], (const float*)d_in[18], (const float*)d_in[26]};
  const float* encWres[3] = {(const float*)d_in[11], (const float*)d_in[19], (const float*)d_in[27]};
  const float* encAe[3]   = {(const float*)d_in[12], (const float*)d_in[20], (const float*)d_in[28]};
  const float* encAl[3]   = {(const float*)d_in[13], (const float*)d_in[21], (const float*)d_in[29]};
  const float* encAr[3]   = {(const float*)d_in[14], (const float*)d_in[22], (const float*)d_in[30]};
  const float* encBeta[3] = {(const float*)d_in[15], (const float*)d_in[23], (const float*)d_in[31]};
  const float* encGamma[3]= {(const float*)d_in[16], (const float*)d_in[24], (const float*)d_in[32]};
  const float* fc1_W = (const float*)d_in[33];
  const float* fc1_b = (const float*)d_in[34];
  const float* fc2_W = (const float*)d_in[35];
  const float* fc2_b = (const float*)d_in[36];
  const float* mask_token = (const float*)d_in[37];
  const float* x         = (const float*)d_in[38];
  const float* edge_attr = (const float*)d_in[39];
  const int* src        = (const int*)d_in[40];
  const int* dst        = (const int*)d_in[41];
  const int* mask_nodes = (const int*)d_in[42];
  const int* pos_idx    = (const int*)d_in[43];
  const int* neg_src    = (const int*)d_in[44];
  const int* neg_dst    = (const int*)d_in[45];

  // workspace layout (floats)
  float* p = (float*)d_ws;
  size_t off = 0;
  float* xm  = p + off; off += (size_t)NN * NDIM;
  float* h1  = p + off; off += (size_t)NN * HID;
  float* h2  = p + off; off += (size_t)NN * HID;
  float* h3  = p + off; off += (size_t)NN * HID;
  float* rep = p + off; off += (size_t)NN * HID;     // later reused as hfc
  float* big = p + off;                               // per-layer edge region, later featcat
  // per-layer sublayout of `big`
  float* feat  = big;                                  // NN*HID (decoder uses NN*64)
  float* recon = feat  + (size_t)NN * HID;             // NN*64
  float* el    = recon + (size_t)NN * NDIM;            // NN*HEADS
  float* er    = el + (size_t)NN * HEADS;
  float* mbuf  = er + (size_t)NN * HEADS;
  float* sbuf  = mbuf + (size_t)NN * HEADS;
  float* logits = sbuf + (size_t)NN * HEADS;           // NE*HEADS (reused as ex)
  size_t big_sz = ((size_t)2 * NSAMP_ * 6 * HID);      // featcat needs the most
  size_t layer_sz = (size_t)(logits + (size_t)NE * HEADS - big);
  if (layer_sz > big_sz) big_sz = layer_sz;
  off += big_sz;
  float* wcomb = p + off; off += EDIM * HEADS;
  float* acc   = p + off; off += 8;
  float* featcat = big;                                // alias after decoder+SCE
  float* hfc = rep;                                    // alias, rep dead by then

  const int T = 256;

  zero_acc<<<1, 32, 0, stream>>>(acc);

  // xm = x with masked rows replaced by mask_token
  copy_f32<<<blks((long)NN * NDIM, T), T, 0, stream>>>(x, xm, (long)NN * NDIM);
  apply_mask_token<<<blks((long)NMASK_ * NDIM, T), T, 0, stream>>>(mask_nodes, mask_token, xm);

  // -------- encoder layers --------
  float* hout[3] = {h1, h2, h3};
  const float* hin = xm;
  int din = NDIM;
  for (int L = 0; L < 3; ++L) {
    make_wcomb<<<1, EDIM * HEADS, 0, stream>>>(encWe[L], HID, encAe[L], HEADS, DHH, wcomb);
    // feat = hin @ W   [NN, din] x [din, HID]
    launch_gemm(hin, din, encW[L], HID, feat, HID, nullptr, nullptr, NN, din, HID, 0, stream);
    // hout = hin @ Wres (residual pre-fill, scatter adds on top)
    launch_gemm(hin, din, encWres[L], HID, hout[L], HID, nullptr, nullptr, NN, din, HID, 0, stream);
    node_attn_init<<<blks((long)NN * HEADS, T), T, 0, stream>>>(
        feat, HID, encAl[L], encAr[L], HEADS, DHH, el, er, mbuf, sbuf, NN);
    edge_logits_max<<<blks((long)NE * HEADS, T), T, 0, stream>>>(
        edge_attr, wcomb, el, er, src, dst, HEADS, logits, mbuf);
    edge_exp_sum<<<blks((long)NE * HEADS, T), T, 0, stream>>>(logits, mbuf, sbuf, dst, HEADS);
    scatter_agg<<<blks((long)NE * (HID / 4), T), T, 0, stream>>>(
        feat, logits, sbuf, src, dst, HEADS, DHH, hout[L]);
    ln_relu<<<blks((long)NN * 32, T), T, 0, stream>>>(hout[L], encGamma[L], encBeta[L], NN);
    hin = hout[L];
    din = HID;
  }

  // rep = concat(h1,h2,h3) @ W_e2d  (three accumulating K-slices)
  launch_gemm(h1, HID, W_e2d + 0 * HID * HID, HID, rep, HID, nullptr, nullptr, NN, HID, HID, 0, stream);
  launch_gemm(h2, HID, W_e2d + 1 * HID * HID, HID, rep, HID, rep, nullptr, NN, HID, HID, 0, stream);
  launch_gemm(h3, HID, W_e2d + 2 * HID * HID, HID, rep, HID, rep, nullptr, NN, HID, HID, 0, stream);

  // -------- decoder GAT (heads=1, dh=64, no LN) --------
  make_wcomb<<<1, EDIM, 0, stream>>>(decWe, NDIM, decAe, 1, NDIM, wcomb);
  launch_gemm(rep, HID, decW, NDIM, feat, NDIM, nullptr, nullptr, NN, HID, NDIM, 0, stream);
  launch_gemm(rep, HID, decWres, NDIM, recon, NDIM, nullptr, nullptr, NN, HID, NDIM, 0, stream);
  node_attn_init<<<blks((long)NN, T), T, 0, stream>>>(
      feat, NDIM, decAl, decAr, 1, NDIM, el, er, mbuf, sbuf, NN);
  edge_logits_max<<<blks((long)NE, T), T, 0, stream>>>(
      edge_attr, wcomb, el, er, src, dst, 1, logits, mbuf);
  edge_exp_sum<<<blks((long)NE, T), T, 0, stream>>>(logits, mbuf, sbuf, dst, 1);
  scatter_agg<<<blks((long)NE * (NDIM / 4), T), T, 0, stream>>>(
      feat, logits, sbuf, src, dst, 1, NDIM, recon);

  // SCE loss on masked nodes (uses recon + original x) -- before featcat aliases `big`
  sce_loss<<<blks((long)NMASK_ * 32, T), T, 0, stream>>>(recon, x, mask_nodes, acc);

  // -------- edge-reconstruction BCE head --------
  gather_featcat<<<blks((long)2 * NSAMP_ * 6 * HID, T), T, 0, stream>>>(
      h1, h2, h3, src, dst, pos_idx, neg_src, neg_dst, featcat);
  // hfc = leakyrelu(featcat @ fc1_W + fc1_b)   [2*NSAMP, 768] x [768, 128]
  launch_gemm(featcat, 6 * HID, fc1_W, HID, hfc, HID, nullptr, fc1_b,
              2 * NSAMP_, 6 * HID, HID, 1, stream);
  bce_loss<<<blks((long)2 * NSAMP_ * 32, T), T, 0, stream>>>(hfc, fc2_W, fc2_b, acc);

  finalize<<<1, 32, 0, stream>>>(acc, (float*)d_out);
}
